// Tripletlosshard2_54125177864861
// MI455X (gfx1250) — matrix operand order repaired and checked by reading, not compiled, not run on hardware
//
#include <hip/hip_runtime.h>
#include <hip/hip_bf16.h>

typedef __attribute__((ext_vector_type(16))) _Float16 v16h;
typedef __attribute__((ext_vector_type(8)))  _Float16 v8h;
typedef __attribute__((ext_vector_type(4)))  _Float16 v4h;
typedef __attribute__((ext_vector_type(8)))  float    v8f;

#define B_ROWS   8192
#define L_ROWS   4096
#define DIM      128
#define TILE_M   16
#define LDSTRIDE 4104   // 4096 + 8 halves of pad per row (row pitch 8208 B, 16B-aligned)

// ---------------------------------------------------------------------------
// Row-normalize f32 -> f16. One wave (32 lanes) per row, 8 rows per block.
// ---------------------------------------------------------------------------
__global__ __launch_bounds__(256) void normalize_f16(const float* __restrict__ in,
                                                     _Float16* __restrict__ out) {
    const int wave = threadIdx.x >> 5;
    const int lane = threadIdx.x & 31;
    const int row  = blockIdx.x * 8 + wave;
    const float4* rp = (const float4*)(in + (size_t)row * DIM);
    float4 v = rp[lane];
    float s = v.x * v.x + v.y * v.y + v.z * v.z + v.w * v.w;
    #pragma unroll
    for (int d = 16; d >= 1; d >>= 1) s += __shfl_xor(s, d, 32);
    const float inv = 1.0f / sqrtf(s);
    v4h h;
    h[0] = (_Float16)(v.x * inv);
    h[1] = (_Float16)(v.y * inv);
    h[2] = (_Float16)(v.z * inv);
    h[3] = (_Float16)(v.w * inv);
    *(v4h*)(out + (size_t)row * DIM + lane * 4) = h;
}

__global__ void zero_accum(float* sums, int* cnts) {
    sums[0] = 0.0f; sums[1] = 0.0f;
    cnts[0] = 0;    cnts[1] = 0;
}

// ---------------------------------------------------------------------------
// Fused kernel: 16-row stripe of sim = tH @ lH^T via WMMA into LDS (f16),
// then per-row/per-level top-2 + masked triplet loss, atomics into ws accums.
// Block = 256 threads = 8 waves; wave w owns columns [w*512, w*512+512).
// B fragments are double-buffered so tile t+1's global loads overlap the
// WMMAs and LDS stores of tile t.
// ---------------------------------------------------------------------------
__global__ __launch_bounds__(256) void triplet_main(
        const _Float16* __restrict__ tH,
        const _Float16* __restrict__ lH,
        const int*      __restrict__ target,
        float* __restrict__ sums,
        int*   __restrict__ cnts) {
    extern __shared__ _Float16 simLds[];   // [TILE_M][LDSTRIDE]

    const int lane    = threadIdx.x & 31;
    const int wave    = threadIdx.x >> 5;
    const int hi      = lane >> 4;         // 0: lanes 0-15, 1: lanes 16-31
    const int l16     = lane & 15;
    const int rowBase = blockIdx.x * TILE_M;

    // ---- Phase 1: WMMA GEMM (16 x 4096, K=128) into LDS --------------------
    // A fragment layout (16-bit A 16x32): lane holds row M=lane&15;
    // lanes 0-15 hold K 0-15, lanes 16-31 hold K 16-31 (matches B).
    v16h aF[4];
    {
        const _Float16* arow = tH + (size_t)(rowBase + l16) * DIM;
        #pragma unroll
        for (int kk = 0; kk < 4; ++kk) {
            union { v16h v; v8h h[2]; } u;
            u.h[0] = *(const v8h*)(arow + kk * 32 + (hi ? 16 : 0));
            u.h[1] = *(const v8h*)(arow + kk * 32 + (hi ? 24 : 8));
            aF[kk] = u.v;
        }
    }
    // B fragment (16-bit B 32x16): lane holds column N=lane&15 (= a label row,
    // since B = labels^T); lanes 0-15 hold K 0-15, lanes 16-31 hold K 16-31.
    const int nWave0 = wave * 512;
    const _Float16* bBase = lH + (size_t)l16 * DIM + (hi ? 16 : 0);

    auto loadB = [&](int t, v16h* dst) {
        const _Float16* brow = bBase + (size_t)(nWave0 + t * 16) * DIM;
        #pragma unroll
        for (int kk = 0; kk < 4; ++kk)
            dst[kk] = *(const v16h*)(brow + kk * 32);
    };

    v16h bF[2][4];
    loadB(0, bF[0]);
    for (int t = 0; t < 32; ++t) {
        const int cur = t & 1;
        if (t + 1 < 32) loadB(t + 1, bF[cur ^ 1]);   // prefetch next tile

        v8f acc = {};
        #pragma unroll
        for (int kk = 0; kk < 4; ++kk) {
            acc = __builtin_amdgcn_wmma_f32_16x16x32_f16(
                      /*neg_a=*/false, aF[kk], /*neg_b=*/false, bF[cur][kk],
                      /*c_mod=*/(short)0, acc, /*reuse_a=*/false, /*reuse_b=*/false);
        }
        // C/D layout: VGPR r -> M = r + 8*hi, N = nBase + l16
        const int m0 = hi << 3;
        const int n  = nWave0 + t * 16 + l16;
        #pragma unroll
        for (int r = 0; r < 8; ++r)
            simLds[(m0 + r) * LDSTRIDE + n] = (_Float16)acc[r];
    }
    __syncthreads();

    // ---- Phase 2+3: per (row, level) top-2 then masked triplet sums --------
    // Lane owns 8 contiguous columns per step: sim read as one ds_load_b128,
    // mask read as two global b128 (coalesced 1 KB per wave-step).
    const int   lvlOff[2] = {0, 1024};
    const int   lvlEnd[2] = {1024, 4096};
    const float lvlMg[2]  = {0.2f, 0.3f};

    #pragma unroll
    for (int rr = 0; rr < 2; ++rr) {
        const int r = wave * 2 + rr;                         // 8 waves x 2 = 16 rows
        const _Float16* srow = simLds + r * LDSTRIDE;
        const int* trow = target + (size_t)(rowBase + r) * (size_t)L_ROWS;

        for (int lvl = 0; lvl < 2; ++lvl) {
            const int   off    = lvlOff[lvl];
            const int   end    = lvlEnd[lvl];
            const float margin = lvlMg[lvl];

            // lane-local top-2 scan (absolute column index), 8 cols per step
            float m1 = -3.4e38f, m2 = -3.4e38f;
            int a1 = -1;
            for (int j0 = off + lane * 8; j0 < end; j0 += 32 * 8) {
                v8h vv = *(const v8h*)(srow + j0);
                #pragma unroll
                for (int e = 0; e < 8; ++e) {
                    float v = (float)vv[e];
                    if (v > m1)      { m2 = m1; m1 = v; a1 = j0 + e; }
                    else if (v > m2) { m2 = v; }
                }
            }
            // wave32 butterfly merge (tie at m1 -> both loss branches give
            // `margin`, so index tie-break is irrelevant)
            #pragma unroll
            for (int d = 16; d >= 1; d >>= 1) {
                float om1 = __shfl_xor(m1, d, 32);
                float om2 = __shfl_xor(m2, d, 32);
                int   oa1 = __shfl_xor(a1, d, 32);
                if (om1 > m1) { m2 = fmaxf(m1, om2); m1 = om1; a1 = oa1; }
                else          { m2 = fmaxf(m2, om1); }
            }

            // masked per-triplet accumulation
            const float tie = fmaxf(m2 - m1 + margin, 0.0f); // j==a1 case
            float ws = 0.0f;
            int   wc = 0;
            for (int j0 = off + lane * 8; j0 < end; j0 += 32 * 8) {
                v8h  vv = *(const v8h*)(srow + j0);
                int4 t0 = *(const int4*)(trow + j0);
                int4 t1 = *(const int4*)(trow + j0 + 4);
                int  tg[8] = {t0.x, t0.y, t0.z, t0.w, t1.x, t1.y, t1.z, t1.w};
                #pragma unroll
                for (int e = 0; e < 8; ++e) {
                    float v   = (float)vv[e];
                    float per = (v < m1) ? (m1 - v + margin)
                              : ((j0 + e == a1) ? tie : margin);
                    ws += per * (float)tg[e];
                    wc += tg[e];
                }
            }
            #pragma unroll
            for (int d = 16; d >= 1; d >>= 1) {
                ws += __shfl_xor(ws, d, 32);
                wc += __shfl_xor(wc, d, 32);
            }
            if (lane == 0) {
                atomicAdd(&sums[lvl], ws);
                atomicAdd(&cnts[lvl], wc);
            }
        }
    }
}

__global__ void finalize_loss(const float* __restrict__ sums,
                              const int*   __restrict__ cnts,
                              float* __restrict__ out) {
    float total = 0.0f;
    #pragma unroll
    for (int l = 0; l < 2; ++l) {
        float c    = (float)cnts[l];
        float loss = sums[l] / fmaxf(c, 1.0f);
        total += (c >= 3.0f) ? loss : 0.0f;
    }
    out[0] = total;
}

// ---------------------------------------------------------------------------
extern "C" void kernel_launch(void* const* d_in, const int* in_sizes, int n_in,
                              void* d_out, int out_size, void* d_ws, size_t ws_size,
                              hipStream_t stream) {
    const float* text   = (const float*)d_in[0];   // (8192, 128) f32
    const float* label  = (const float*)d_in[1];   // (4096, 128) f32
    const int*   target = (const int*)d_in[2];     // (8192, 4096) i32

    char* ws = (char*)d_ws;
    _Float16* tH = (_Float16*)ws;                                    // 2 MB
    _Float16* lH = (_Float16*)(ws + (size_t)B_ROWS * DIM * 2);       // 1 MB
    float* sums  = (float*)(ws + (size_t)B_ROWS * DIM * 2
                               + (size_t)L_ROWS * DIM * 2);          // 8 B
    int*   cnts  = (int*)(sums + 2);                                 // 8 B

    normalize_f16<<<B_ROWS / 8, 256, 0, stream>>>(text,  tH);
    normalize_f16<<<L_ROWS / 8, 256, 0, stream>>>(label, lH);
    zero_accum<<<1, 1, 0, stream>>>(sums, cnts);

    const size_t ldsBytes = (size_t)TILE_M * LDSTRIDE * sizeof(_Float16); // 131,328 B
    (void)hipFuncSetAttribute((const void*)triplet_main,
                              hipFuncAttributeMaxDynamicSharedMemorySize,
                              (int)ldsBytes);
    triplet_main<<<B_ROWS / TILE_M, 256, ldsBytes, stream>>>(tH, lH, target,
                                                             sums, cnts);
    finalize_loss<<<1, 1, 0, stream>>>(sums, cnts, (float*)d_out);
}
